// MultiLatentHeadAttn_62964220559621
// MI455X (gfx1250) — matrix operand order repaired and checked by reading, not compiled, not run on hardware
//
#include <hip/hip_runtime.h>

typedef __bf16 bf16_t;
typedef unsigned int u32;
typedef __attribute__((ext_vector_type(16))) __bf16 v16bf;
typedef __attribute__((ext_vector_type(8)))  __bf16 v8bf;
typedef __attribute__((ext_vector_type(8)))  float  v8f;

// Problem constants
#define BB 2
#define TT 2048
#define DMODEL 2048
#define NHEAD 16
#define DCC 512
#define DHEAD 128
#define DQKDIM 160   // 128 content + 32 rope

// ---------------------------------------------------------------------------
// Fragment loader: 16 bf16 as two 16-byte vector loads.
// ---------------------------------------------------------------------------
__device__ __forceinline__ v16bf ld16(const bf16_t* p, int hi_off) {
  v8bf lo = *(const v8bf*)p;
  v8bf hi = *(const v8bf*)(p + hi_off);
  return __builtin_shufflevector(lo, hi, 0, 1, 2, 3, 4, 5, 6, 7,
                                 8, 9, 10, 11, 12, 13, 14, 15);
}

__device__ __forceinline__ v8f wmma_bf16(v16bf a, v16bf b, v8f c) {
  return __builtin_amdgcn_wmma_f32_16x16x32_bf16(false, a, false, b,
                                                 (short)0, c, false, false);
}

// ---------------------------------------------------------------------------
// fp32 -> bf16 conversion
// ---------------------------------------------------------------------------
__global__ void cvt_f32_bf16(const float* __restrict__ in, bf16_t* __restrict__ out, int n) {
  int i = blockIdx.x * blockDim.x + threadIdx.x;
  if (i < n) out[i] = (bf16_t)in[i];
}

// ---------------------------------------------------------------------------
// bf16 GEMM: C[M,N] = A[M,K] @ W[K,N] + bias[N], bf16 out, fp32 accum.
// Block tile 128x128, K-step 32, double-buffered LDS, register-staged
// pipeline: global loads for tile s+1 issue BEFORE the WMMAs of tile s,
// LDS stores land after (loadcnt wait hidden behind matrix math).
// ---------------------------------------------------------------------------
#define ASTR 40   // 32 + pad (multiple of 8 -> 16B-aligned rows)
#define TSTR 40   // transposed tile stride

__global__ __launch_bounds__(256) void gemm_bf16_wmma(
    const bf16_t* __restrict__ A, int lda,
    const bf16_t* __restrict__ W, int ldb,
    const float*  __restrict__ bias,
    bf16_t* __restrict__ C, int ldc,
    int M, int N, int K) {
  __shared__ __attribute__((aligned(16))) bf16_t As[2][128 * ASTR];
  __shared__ __attribute__((aligned(16))) bf16_t BsT[2][128 * TSTR];

  const int tid  = threadIdx.x;
  const int lane = tid & 31;
  const int wave = tid >> 5;
  const int wm   = wave & 3;          // M group (32 rows)
  const int wn   = wave >> 2;         // N group (64 cols)
  const int l16  = lane & 15;
  const int lh   = lane >> 4;
  const int bm   = blockIdx.x * 128;
  const int bn   = blockIdx.y * 128;

  // Per-thread staging coordinates (fixed)
  const int ar  = tid >> 2;           // A row base (+64 for i=1)
  const int ac  = (tid & 3) * 8;      // A col chunk
  const int bnp = (tid & 63) * 2;     // W col pair
  const int bkp = (tid >> 6) * 2;     // W row pair base (+8 per i)
  const bf16_t* Ag = A + (size_t)(bm + ar) * lda + ac;
  const bf16_t* Wg = W + (size_t)bkp * ldb + bn + bnp;

  v8bf atmp[2];
  u32  btmp[8];

  auto load_next = [&](int k0) {       // global -> registers (no LDS yet)
    const bf16_t* ap = Ag + k0;
#pragma unroll
    for (int i = 0; i < 2; ++i)
      atmp[i] = *(const v8bf*)(ap + (size_t)(i * 64) * lda);
    const bf16_t* bp = Wg + (size_t)k0 * ldb;
#pragma unroll
    for (int i = 0; i < 4; ++i) {
      const bf16_t* s = bp + (size_t)(i * 8) * ldb;
      btmp[2 * i]     = *(const u32*)s;
      btmp[2 * i + 1] = *(const u32*)(s + ldb);
    }
  };
  auto store_next = [&](int buf) {     // registers -> LDS (2x2 transpose for W)
#pragma unroll
    for (int i = 0; i < 2; ++i)
      *(v8bf*)&As[buf][(ar + i * 64) * ASTR + ac] = atmp[i];
#pragma unroll
    for (int i = 0; i < 4; ++i) {
      u32 a = btmp[2 * i], b = btmp[2 * i + 1];
      u32 lo = (a & 0xffffu) | (b << 16);         // {W[k][n],   W[k+1][n]}
      u32 hi = (a >> 16)     | (b & 0xffff0000u);  // {W[k][n+1], W[k+1][n+1]}
      *(u32*)&BsT[buf][(bnp)     * TSTR + bkp + i * 8] = lo;
      *(u32*)&BsT[buf][(bnp + 1) * TSTR + bkp + i * 8] = hi;
    }
  };

  v8f acc[2][4] = {};

  const int nsteps = K >> 5;
  load_next(0);
  store_next(0);
  __syncthreads();

  for (int s = 0; s < nsteps; ++s) {
    const int cur = s & 1, nxt = cur ^ 1;
    const bool more = (s + 1 < nsteps);
    if (more) load_next((s + 1) * 32);   // loads in flight during WMMAs

    v16bf af0 = ld16(&As[cur][(wm * 32 + l16) * ASTR + lh * 8], 16);
    v16bf af1 = ld16(&As[cur][(wm * 32 + 16 + l16) * ASTR + lh * 8], 16);
#pragma unroll
    for (int nf = 0; nf < 4; ++nf) {
      int col = wn * 64 + nf * 16 + l16;
      v16bf bf = ld16(&BsT[cur][col * TSTR + lh * 8], 16);   // K = lh*8 + e (e<8), 16 + lh*8 + (e-8) (e>=8)
      acc[0][nf] = wmma_bf16(af0, bf, acc[0][nf]);
      acc[1][nf] = wmma_bf16(af1, bf, acc[1][nf]);
    }
    if (more) store_next(nxt);           // loadcnt wait lands here
    __syncthreads();
  }

  // Epilogue: bias add, convert to bf16, scatter-store
#pragma unroll
  for (int mf = 0; mf < 2; ++mf) {
#pragma unroll
    for (int nf = 0; nf < 4; ++nf) {
      int col = bn + wn * 64 + nf * 16 + l16;
      float bv = bias[col];
#pragma unroll
      for (int r = 0; r < 8; ++r) {
        int row = bm + wm * 32 + mf * 16 + lh * 8 + r;
        C[(size_t)row * ldc + col] = (bf16_t)(acc[mf][nf][r] + bv);
      }
    }
  }
}

// ---------------------------------------------------------------------------
// Assemble Q or K (with optional fold-in of the softmax scale):
//   dst[b,h,t,0:128]   = scale * content[b*T+t, h*128+:]
//   dst[b,h,t,128:160] = scale * RoPE(rope_pre[b*T+t, h*32+:])
// ---------------------------------------------------------------------------
__global__ void assemble_qk(const bf16_t* __restrict__ content,  // [B*T, 2048]
                            const bf16_t* __restrict__ ropein,   // [B*T, 512]
                            bf16_t* __restrict__ dst,            // [B,H,T,160]
                            float scale) {
  long long idx = (long long)blockIdx.x * blockDim.x + threadIdx.x;
  const long long total = (long long)BB * NHEAD * TT * DQKDIM;
  if (idx >= total) return;
  int d = (int)(idx % DQKDIM);
  long long tmp = idx / DQKDIM;
  int t = (int)(tmp % TT); tmp /= TT;
  int h = (int)(tmp % NHEAD);
  int b = (int)(tmp / NHEAD);
  long long row = (long long)b * TT + t;

  float val;
  if (d < DHEAD) {
    val = (float)content[row * DMODEL + h * DHEAD + d];
  } else {
    int j = d - DHEAD;          // 0..31
    int i = (j < 16) ? j : (j - 16);
    float fr = powf(10000.0f, -(2.0f * i) / 32.0f);
    float th = (float)t * fr;
    float c = __cosf(th), s = __sinf(th);
    float x1 = (float)ropein[row * DCC + h * 32 + 2 * i];
    float x2 = (float)ropein[row * DCC + h * 32 + 2 * i + 1];
    val = (j < 16) ? (x1 * c - x2 * s) : (x1 * s + x2 * c);
  }
  dst[idx] = (bf16_t)(val * scale);
}

// ---------------------------------------------------------------------------
// Flash attention (no mask). 256 thr = 8 waves; each wave owns 16 Q rows and
// the full 128 output dims; 32 keys/iteration; K/V double-buffered in LDS
// with register-staged pipeline. Softmax scale pre-folded into Q.
// ---------------------------------------------------------------------------
#define KSTR 168   // 160 + pad (multiple of 8)
#define PSTR 40    // 32 + pad

__global__ __launch_bounds__(256) void mla_attention(
    const bf16_t* __restrict__ Q,   // [B,H,T,160] (pre-scaled)
    const bf16_t* __restrict__ Kg,  // [B,H,T,160]
    const bf16_t* __restrict__ Vf,  // [B*T, 2048]
    float* __restrict__ out) {      // [B,T,2048]
  __shared__ __attribute__((aligned(16))) bf16_t Ks[2][32 * KSTR];
  __shared__ __attribute__((aligned(16))) bf16_t VsT[2][128 * TSTR];
  __shared__ __attribute__((aligned(16))) bf16_t Ps[8 * 16 * PSTR];

  const int tid  = threadIdx.x;
  const int lane = tid & 31;
  const int wave = tid >> 5;
  const int l16  = lane & 15;
  const int lh   = lane >> 4;
  const int h    = blockIdx.y;
  const int bz   = blockIdx.z;
  const int q0   = blockIdx.x * 128 + wave * 16;

  const long long bh = (long long)bz * NHEAD + h;
  const bf16_t* Qbase = Q  + (bh * TT + q0) * DQKDIM;
  const bf16_t* Kbase = Kg + bh * TT * DQKDIM;
  const bf16_t* Vbase = Vf + (long long)bz * TT * DMODEL + h * DHEAD;

  v8bf ktmp[3];
  u32  vtmp[8];

  auto loadKV = [&](int key0) {        // global -> registers
    const bf16_t* kp = Kbase + (size_t)key0 * DQKDIM;
#pragma unroll
    for (int i = 0; i < 3; ++i) {
      int lin = tid + i * 256;
      if (lin < 640) {
        int r = lin / 20, c8 = (lin % 20) * 8;
        ktmp[i] = *(const v8bf*)(kp + r * DQKDIM + c8);
      }
    }
    const bf16_t* vp = Vbase + (size_t)key0 * DMODEL;
#pragma unroll
    for (int i = 0; i < 4; ++i) {
      int lin = tid + i * 256;
      int np = (lin & 63) * 2;
      int k2 = (lin >> 6) * 2;
      const bf16_t* s = vp + (size_t)k2 * DMODEL + np;
      vtmp[2 * i]     = *(const u32*)s;
      vtmp[2 * i + 1] = *(const u32*)(s + DMODEL);
    }
  };
  auto storeKV = [&](int buf) {        // registers -> LDS
#pragma unroll
    for (int i = 0; i < 3; ++i) {
      int lin = tid + i * 256;
      if (lin < 640) {
        int r = lin / 20, c8 = (lin % 20) * 8;
        *(v8bf*)&Ks[buf][r * KSTR + c8] = ktmp[i];
      }
    }
#pragma unroll
    for (int i = 0; i < 4; ++i) {
      int lin = tid + i * 256;
      int np = (lin & 63) * 2;
      int k2 = (lin >> 6) * 2;
      u32 a = vtmp[2 * i], b = vtmp[2 * i + 1];
      u32 lo = (a & 0xffffu) | (b << 16);
      u32 hi = (a >> 16)     | (b & 0xffff0000u);
      *(u32*)&VsT[buf][(np)     * TSTR + k2] = lo;
      *(u32*)&VsT[buf][(np + 1) * TSTR + k2] = hi;
    }
  };

  // Resident Q fragments: 5 K-chunks of 32 (two b128 global loads each)
  v16bf qa[5];
#pragma unroll
  for (int c = 0; c < 5; ++c)
    qa[c] = ld16(Qbase + (long long)l16 * DQKDIM + c * 32 + lh * 8, 16);

  v8f o[8] = {};
  float m_i[8], l_i[8];
#pragma unroll
  for (int r = 0; r < 8; ++r) { m_i[r] = -3.0e38f; l_i[r] = 0.0f; }

  const int nT = TT / 32;
  loadKV(0);
  storeKV(0);
  __syncthreads();

  for (int kt = 0; kt < nT; ++kt) {
    const int cur = kt & 1, nxt = cur ^ 1;
    const int key0 = kt * 32;
    const bool more = (kt + 1 < nT);
    if (more) loadKV(key0 + 32);       // loads in flight across S+softmax+PV
    if (kt + 2 < nT)
      __builtin_prefetch(&Kbase[(long long)(key0 + 64 + lane) * DQKDIM], 0, 1);

    // S = Q K^T for 32 keys: two 16x16 fragments
    v8f s0 = {}, s1 = {};
#pragma unroll
    for (int c = 0; c < 5; ++c) {
      v16bf b0 = ld16(&Ks[cur][l16 * KSTR + c * 32 + lh * 8], 16);
      v16bf b1 = ld16(&Ks[cur][(16 + l16) * KSTR + c * 32 + lh * 8], 16);
      s0 = wmma_bf16(qa[c], b0, s0);
      s1 = wmma_bf16(qa[c], b1, s1);
    }

    // Online softmax (per lane: 8 rows, 1 col per fragment)
    float alpha[8];
#pragma unroll
    for (int r = 0; r < 8; ++r) {
      float v0 = s0[r];
      float v1 = s1[r];
      float tmax = fmaxf(v0, v1);
#pragma unroll
      for (int off = 1; off < 16; off <<= 1)
        tmax = fmaxf(tmax, __shfl_xor(tmax, off, 16));
      float mnew = fmaxf(m_i[r], tmax);
      float p0 = __expf(v0 - mnew);
      float p1 = __expf(v1 - mnew);
      float rs = p0 + p1;
#pragma unroll
      for (int off = 1; off < 16; off <<= 1)
        rs += __shfl_xor(rs, off, 16);
      alpha[r] = __expf(m_i[r] - mnew);
      l_i[r] = l_i[r] * alpha[r] + rs;
      m_i[r] = mnew;
      // Stage probabilities: C-layout -> wave-private LDS tile (16x32)
      Ps[(wave * 16 + lh * 8 + r) * PSTR + l16]      = (bf16_t)p0;
      Ps[(wave * 16 + lh * 8 + r) * PSTR + 16 + l16] = (bf16_t)p1;
    }
    // Rescale accumulators
#pragma unroll
    for (int f = 0; f < 8; ++f)
#pragma unroll
      for (int r = 0; r < 8; ++r)
        o[f][r] *= alpha[r];
    __syncthreads();   // (A) Ps visible

    // P in A-matrix layout; O += P @ V (8 column fragments from VsT)
    v16bf pa = ld16(&Ps[(wave * 16 + l16) * PSTR + lh * 8], 16);
#pragma unroll
    for (int f = 0; f < 8; ++f) {
      v16bf bv = ld16(&VsT[cur][(f * 16 + l16) * TSTR + lh * 8], 16);
      o[f] = wmma_bf16(pa, bv, o[f]);
    }
    if (more) storeKV(nxt);            // loadcnt wait lands here
    __syncthreads();   // (B) nxt staged; cur free next iteration
  }

  // Normalize and store fp32 output: out[b, t, h*128 + d]
  float inv_l[8];
#pragma unroll
  for (int r = 0; r < 8; ++r) inv_l[r] = 1.0f / l_i[r];
#pragma unroll
  for (int f = 0; f < 8; ++f) {
    int col = h * DHEAD + f * 16 + l16;
#pragma unroll
    for (int r = 0; r < 8; ++r) {
      int row = q0 + lh * 8 + r;
      out[((long long)bz * TT + row) * DMODEL + col] = o[f][r] * inv_l[r];
    }
  }
}

// ---------------------------------------------------------------------------
// Host launcher
// ---------------------------------------------------------------------------
extern "C" void kernel_launch(void* const* d_in, const int* in_sizes, int n_in,
                              void* d_out, int out_size, void* d_ws, size_t ws_size,
                              hipStream_t stream) {
  const float* x     = (const float*)d_in[0];
  const float* W_DKV = (const float*)d_in[1];
  const float* b_DKV = (const float*)d_in[2];
  const float* W_UK  = (const float*)d_in[3];
  const float* b_UK  = (const float*)d_in[4];
  const float* W_UV  = (const float*)d_in[5];
  const float* b_UV  = (const float*)d_in[6];
  const float* W_DQ  = (const float*)d_in[7];
  const float* b_DQ  = (const float*)d_in[8];
  const float* W_UQ  = (const float*)d_in[9];
  const float* b_UQ  = (const float*)d_in[10];
  const float* W_QR  = (const float*)d_in[11];
  const float* b_QR  = (const float*)d_in[12];
  const float* W_KR  = (const float*)d_in[13];
  const float* b_KR  = (const float*)d_in[14];

  const int M = BB * TT;  // 4096 tokens

  char* ws = (char*)d_ws;
  size_t off = 0;
  auto alloc = [&](size_t elems) -> bf16_t* {
    bf16_t* p = (bf16_t*)(ws + off);
    off = (off + elems * sizeof(bf16_t) + 255) & ~(size_t)255;
    return p;
  };

  bf16_t* xb   = alloc((size_t)M * DMODEL);
  bf16_t* wdkv = alloc((size_t)DMODEL * DCC);
  bf16_t* wuk  = alloc((size_t)DCC * DMODEL);
  bf16_t* wuv  = alloc((size_t)DCC * DMODEL);
  bf16_t* wdq  = alloc((size_t)DMODEL * DCC);
  bf16_t* wuq  = alloc((size_t)DCC * DMODEL);
  bf16_t* wqr  = alloc((size_t)DCC * DCC);
  bf16_t* wkr  = alloc((size_t)DMODEL * DCC);
  bf16_t* ckv  = alloc((size_t)M * DCC);
  bf16_t* cq   = alloc((size_t)M * DCC);
  bf16_t* kc   = alloc((size_t)M * DMODEL);
  bf16_t* vb   = alloc((size_t)M * DMODEL);
  bf16_t* qc   = alloc((size_t)M * DMODEL);
  bf16_t* qr   = alloc((size_t)M * DCC);
  bf16_t* kr   = alloc((size_t)M * DCC);
  bf16_t* Qb   = alloc((size_t)BB * NHEAD * TT * DQKDIM);
  bf16_t* Kb   = alloc((size_t)BB * NHEAD * TT * DQKDIM);

  auto cvt = [&](const float* src, bf16_t* dst, int n) {
    cvt_f32_bf16<<<(n + 255) / 256, 256, 0, stream>>>(src, dst, n);
  };
  cvt(x, xb, M * DMODEL);
  cvt(W_DKV, wdkv, DMODEL * DCC);
  cvt(W_UK, wuk, DCC * DMODEL);
  cvt(W_UV, wuv, DCC * DMODEL);
  cvt(W_DQ, wdq, DMODEL * DCC);
  cvt(W_UQ, wuq, DCC * DMODEL);
  cvt(W_QR, wqr, DCC * DCC);
  cvt(W_KR, wkr, DMODEL * DCC);

  auto gemm = [&](const bf16_t* A, int lda, const bf16_t* W, int ldb,
                  const float* bias, bf16_t* C, int ldc, int m, int n, int k) {
    gemm_bf16_wmma<<<dim3(m / 128, n / 128), 256, 0, stream>>>(
        A, lda, W, ldb, bias, C, ldc, m, n, k);
  };
  gemm(xb,  DMODEL, wdkv, DCC,    b_DKV, ckv, DCC,    M, DCC,    DMODEL);  // c_kv
  gemm(ckv, DCC,    wuk,  DMODEL, b_UK,  kc,  DMODEL, M, DMODEL, DCC);     // k_content
  gemm(ckv, DCC,    wuv,  DMODEL, b_UV,  vb,  DMODEL, M, DMODEL, DCC);     // v
  gemm(xb,  DMODEL, wdq,  DCC,    b_DQ,  cq,  DCC,    M, DCC,    DMODEL);  // c_q
  gemm(cq,  DCC,    wuq,  DMODEL, b_UQ,  qc,  DMODEL, M, DMODEL, DCC);     // q_content
  gemm(cq,  DCC,    wqr,  DCC,    b_QR,  qr,  DCC,    M, DCC,    DCC);     // q_rope pre
  gemm(xb,  DMODEL, wkr,  DCC,    b_KR,  kr,  DCC,    M, DCC,    DMODEL);  // k_rope pre

  long long nq = (long long)BB * NHEAD * TT * DQKDIM;
  int blocks = (int)((nq + 255) / 256);
  const float scale = 1.0f / sqrtf((float)DQKDIM);
  assemble_qk<<<blocks, 256, 0, stream>>>(qc, qr, Qb, scale);  // Q pre-scaled
  assemble_qk<<<blocks, 256, 0, stream>>>(kc, kr, Kb, 1.0f);   // K unscaled

  mla_attention<<<dim3(TT / 128, NHEAD, BB), 256, 0, stream>>>(
      Qb, Kb, vb, (float*)d_out);
}
